// GraphWeatherProcessor_15006615733527
// MI455X (gfx1250) — matrix operand order repaired and checked by reading, not compiled, not run on hardware
//
#include <hip/hip_runtime.h>

typedef __attribute__((ext_vector_type(16))) _Float16 v16h;
typedef __attribute__((ext_vector_type(8)))  _Float16 v8h;
typedef __attribute__((ext_vector_type(8)))  float    v8f;
typedef _Float16 half_t;

#define HID 128
#define WPB 2   // waves per block (wave32); each wave does 2 M-tiles (32 rows)

__device__ __forceinline__ float4 ld4(const float* p) { return *(const float4*)p; }

// A-fragment from f16 row pointer p (points at K-offset kk*32 + hi*8):
// elements 0..7 = p[0..7], elements 8..15 = p[16..23]
__device__ __forceinline__ v16h ld_a_h(const half_t* p) {
    v8h lo = *(const v8h*)(p);
    v8h hh = *(const v8h*)(p + 16);
    return __builtin_shufflevector(lo, hh, 0,1,2,3,4,5,6,7,8,9,10,11,12,13,14,15);
}

// A-fragment from f32 LDS row pointer
__device__ __forceinline__ v16h ld_a_f(const float* p) {
    float4 f0 = ld4(p), f1 = ld4(p + 4), f2 = ld4(p + 16), f3 = ld4(p + 20);
    v16h a;
    a[0]=(half_t)f0.x; a[1]=(half_t)f0.y; a[2]=(half_t)f0.z; a[3]=(half_t)f0.w;
    a[4]=(half_t)f1.x; a[5]=(half_t)f1.y; a[6]=(half_t)f1.z; a[7]=(half_t)f1.w;
    a[8]=(half_t)f2.x; a[9]=(half_t)f2.y; a[10]=(half_t)f2.z; a[11]=(half_t)f2.w;
    a[12]=(half_t)f3.x; a[13]=(half_t)f3.y; a[14]=(half_t)f3.z; a[15]=(half_t)f3.w;
    return a;
}

// bias + LayerNorm + ReLU on one 16x128 C-tile held as acc[8] (v8f each)
__device__ __forceinline__ void bias_ln_relu(v8f* acc, const float* b1c,
                                             const float* gc, const float* blc) {
    #pragma unroll
    for (int v = 0; v < 8; ++v) {
        float s1 = 0.f, s2 = 0.f;
        #pragma unroll
        for (int t = 0; t < 8; ++t) {
            float val = acc[t][v] + b1c[t];
            acc[t][v] = val; s1 += val; s2 += val * val;
        }
        #pragma unroll
        for (int m = 1; m < 16; m <<= 1) {
            s1 += __shfl_xor(s1, m, 32);
            s2 += __shfl_xor(s2, m, 32);
        }
        float mean = s1 * (1.0f / HID);
        float var  = s2 * (1.0f / HID) - mean * mean;
        float rstd = rsqrtf(var + 1e-5f);
        #pragma unroll
        for (int t = 0; t < 8; ++t)
            acc[t][v] = fmaxf((acc[t][v] - mean) * rstd * gc[t] + blc[t], 0.f);
    }
}

// Row-major KxHID fp32 weight -> per-lane f16 B-fragment layout
__global__ void swizzle_weights(const float* __restrict__ W, half_t* __restrict__ out, int K) {
    int tid = blockIdx.x * blockDim.x + threadIdx.x;
    if (tid >= K * HID) return;
    int j    = tid & 15;
    int lane = (tid >> 4) & 31;
    int t    = (tid >> 9) & 7;
    int kk   = tid >> 12;
    int krow = kk * 32 + ((j >> 3) << 4) + ((lane >> 4) << 3) + (j & 7);
    int col  = t * 16 + (lane & 15);
    out[tid] = (half_t)W[krow * HID + col];
}

__global__ void cvt_f32_to_f16(const float* __restrict__ in, half_t* __restrict__ out, long n) {
    long i = ((long)blockIdx.x * blockDim.x + threadIdx.x) * 4;
    if (i + 3 < n) {
        float4 f = ld4(in + i);
        v8h* dummy; (void)dummy;
        out[i]   = (half_t)f.x; out[i+1] = (half_t)f.y;
        out[i+2] = (half_t)f.z; out[i+3] = (half_t)f.w;
    } else {
        for (; i < n; ++i) out[i] = (half_t)in[i];
    }
}

// ---------------- message kernel: per-edge MLP + scatter-add ----------------
__global__ __launch_bounds__(64) void msg_kernel(
    const half_t* __restrict__ xh, const int* __restrict__ ei,
    const half_t* __restrict__ w1f, const half_t* __restrict__ w2f,
    const float* __restrict__ b1, const float* __restrict__ gln,
    const float* __restrict__ bln, const float* __restrict__ b2,
    float* __restrict__ aggr, long E, long R)
{
    __shared__ float lds_h[WPB][2][16 * HID];
    const int wave = threadIdx.x >> 5;
    const int lane = threadIdx.x & 31;
    const long base = ((long)blockIdx.x * WPB + wave) * 32;
    if (base >= R) return;
    const int hi   = lane >> 4;
    const int lcol = lane & 15;
    const int kb   = hi * 8;

    long ar0 = base + lcol;            if (ar0 >= R) ar0 = R - 1;
    long ar1 = base + 16 + lcol;       if (ar1 >= R) ar1 = R - 1;
    int s0, d0, s1, d1;
    if (ar0 < E) { s0 = ei[ar0]; d0 = ei[E + ar0]; } else { s0 = d0 = (int)(ar0 - E); }
    if (ar1 < E) { s1 = ei[ar1]; d1 = ei[E + ar1]; } else { s1 = d1 = (int)(ar1 - E); }

    v8f acc0[8] = {}, acc1[8] = {};
    // GEMM1: [32 x 256] x [256 x 128], B-fragment shared by both M-tiles
    #pragma unroll
    for (int kk = 0; kk < 8; ++kk) {
        int r0 = (kk < 4) ? d0 : s0;
        int r1 = (kk < 4) ? d1 : s1;
        int ko = ((kk < 4) ? kk : kk - 4) * 32 + kb;
        v16h a0 = ld_a_h(xh + (long)r0 * HID + ko);
        v16h a1 = ld_a_h(xh + (long)r1 * HID + ko);
        #pragma unroll
        for (int t = 0; t < 8; ++t) {
            v16h b = *(const v16h*)(w1f + (((kk * 8 + t) * 32 + lane) << 4));
            acc0[t] = __builtin_amdgcn_wmma_f32_16x16x32_f16(false, a0, false, b,
                                                             (short)0, acc0[t], false, false);
            acc1[t] = __builtin_amdgcn_wmma_f32_16x16x32_f16(false, a1, false, b,
                                                             (short)0, acc1[t], false, false);
        }
    }
    float b1c[8], gc[8], blc[8], b2c[8];
    #pragma unroll
    for (int t = 0; t < 8; ++t) {
        int col = lcol + 16 * t;
        b1c[t] = b1[col]; gc[t] = gln[col]; blc[t] = bln[col]; b2c[t] = b2[col];
    }
    bias_ln_relu(acc0, b1c, gc, blc);
    bias_ln_relu(acc1, b1c, gc, blc);

    // C-layout -> A-layout transpose via per-wave LDS (f32)
    #pragma unroll
    for (int t = 0; t < 8; ++t) {
        int col = lcol + 16 * t;
        #pragma unroll
        for (int v = 0; v < 8; ++v) {
            int r = hi ? v + 8 : v;
            lds_h[wave][0][r * HID + col] = acc0[t][v];
            lds_h[wave][1][r * HID + col] = acc1[t][v];
        }
    }
    asm volatile("s_wait_dscnt 0" ::: "memory");

    v8f o0[8] = {}, o1[8] = {};
    // GEMM2: [32 x 128] x [128 x 128]
    #pragma unroll
    for (int kk = 0; kk < 4; ++kk) {
        v16h a0 = ld_a_f(&lds_h[wave][0][lcol * HID + kk * 32 + kb]);
        v16h a1 = ld_a_f(&lds_h[wave][1][lcol * HID + kk * 32 + kb]);
        #pragma unroll
        for (int t = 0; t < 8; ++t) {
            v16h b = *(const v16h*)(w2f + (((kk * 8 + t) * 32 + lane) << 4));
            o0[t] = __builtin_amdgcn_wmma_f32_16x16x32_f16(false, a0, false, b,
                                                           (short)0, o0[t], false, false);
            o1[t] = __builtin_amdgcn_wmma_f32_16x16x32_f16(false, a1, false, b,
                                                           (short)0, o1[t], false, false);
        }
    }
    // scatter-add into aggr[dst]
    #pragma unroll
    for (int v = 0; v < 8; ++v) {
        int r = hi ? v + 8 : v;
        if (base + r < R) {
            int dv = __shfl(d0, r, 32);
            float* ap = aggr + (long)dv * HID;
            #pragma unroll
            for (int t = 0; t < 8; ++t)
                atomicAdd(ap + lcol + 16 * t, o0[t][v] + b2c[t]);
        }
        if (base + 16 + r < R) {
            int dv = __shfl(d1, r, 32);
            float* ap = aggr + (long)dv * HID;
            #pragma unroll
            for (int t = 0; t < 8; ++t)
                atomicAdd(ap + lcol + 16 * t, o1[t][v] + b2c[t]);
        }
    }
}

// ---------------- update kernel: per-node MLP ----------------
__global__ __launch_bounds__(64) void upd_kernel(
    const half_t* __restrict__ xh, const half_t* __restrict__ aggrh,
    const half_t* __restrict__ w1f, const half_t* __restrict__ w2f,
    const float* __restrict__ b1, const float* __restrict__ gln,
    const float* __restrict__ bln, const float* __restrict__ b2,
    half_t* __restrict__ xout, float* __restrict__ outf, long Nn)
{
    __shared__ float lds_h[WPB][2][16 * HID];
    const int wave = threadIdx.x >> 5;
    const int lane = threadIdx.x & 31;
    const long base = ((long)blockIdx.x * WPB + wave) * 32;
    if (base >= Nn) return;
    const int hi   = lane >> 4;
    const int lcol = lane & 15;
    const int kb   = hi * 8;

    long ar0 = base + lcol;       if (ar0 >= Nn) ar0 = Nn - 1;
    long ar1 = base + 16 + lcol;  if (ar1 >= Nn) ar1 = Nn - 1;

    v8f acc0[8] = {}, acc1[8] = {};
    #pragma unroll
    for (int kk = 0; kk < 8; ++kk) {
        const half_t* src = (kk < 4) ? xh : aggrh;
        int ko = ((kk < 4) ? kk : kk - 4) * 32 + kb;
        v16h a0 = ld_a_h(src + ar0 * HID + ko);
        v16h a1 = ld_a_h(src + ar1 * HID + ko);
        #pragma unroll
        for (int t = 0; t < 8; ++t) {
            v16h b = *(const v16h*)(w1f + (((kk * 8 + t) * 32 + lane) << 4));
            acc0[t] = __builtin_amdgcn_wmma_f32_16x16x32_f16(false, a0, false, b,
                                                             (short)0, acc0[t], false, false);
            acc1[t] = __builtin_amdgcn_wmma_f32_16x16x32_f16(false, a1, false, b,
                                                             (short)0, acc1[t], false, false);
        }
    }
    float b1c[8], gc[8], blc[8], b2c[8];
    #pragma unroll
    for (int t = 0; t < 8; ++t) {
        int col = lcol + 16 * t;
        b1c[t] = b1[col]; gc[t] = gln[col]; blc[t] = bln[col]; b2c[t] = b2[col];
    }
    bias_ln_relu(acc0, b1c, gc, blc);
    bias_ln_relu(acc1, b1c, gc, blc);

    #pragma unroll
    for (int t = 0; t < 8; ++t) {
        int col = lcol + 16 * t;
        #pragma unroll
        for (int v = 0; v < 8; ++v) {
            int r = hi ? v + 8 : v;
            lds_h[wave][0][r * HID + col] = acc0[t][v];
            lds_h[wave][1][r * HID + col] = acc1[t][v];
        }
    }
    asm volatile("s_wait_dscnt 0" ::: "memory");

    v8f o0[8] = {}, o1[8] = {};
    #pragma unroll
    for (int kk = 0; kk < 4; ++kk) {
        v16h a0 = ld_a_f(&lds_h[wave][0][lcol * HID + kk * 32 + kb]);
        v16h a1 = ld_a_f(&lds_h[wave][1][lcol * HID + kk * 32 + kb]);
        #pragma unroll
        for (int t = 0; t < 8; ++t) {
            v16h b = *(const v16h*)(w2f + (((kk * 8 + t) * 32 + lane) << 4));
            o0[t] = __builtin_amdgcn_wmma_f32_16x16x32_f16(false, a0, false, b,
                                                           (short)0, o0[t], false, false);
            o1[t] = __builtin_amdgcn_wmma_f32_16x16x32_f16(false, a1, false, b,
                                                           (short)0, o1[t], false, false);
        }
    }
    #pragma unroll
    for (int v = 0; v < 8; ++v) {
        int r = hi ? v + 8 : v;
        #pragma unroll
        for (int m = 0; m < 2; ++m) {
            long grow = base + 16 * m + r;
            if (grow >= Nn) continue;
            v8f* o = m ? o1 : o0;
            #pragma unroll
            for (int t = 0; t < 8; ++t) {
                float val = o[t][v] + b2c[t];
                xout[grow * HID + lcol + 16 * t] = (half_t)val;
                if (outf) outf[grow * HID + lcol + 16 * t] = val;
            }
        }
    }
}

extern "C" void kernel_launch(void* const* d_in, const int* in_sizes, int n_in,
                              void* d_out, int out_size, void* d_ws, size_t ws_size,
                              hipStream_t stream) {
    (void)n_in; (void)out_size; (void)ws_size;
    const float* x_in = (const float*)d_in[0];
    const int*   ei   = (const int*)d_in[1];
    const float* Wm1  = (const float*)d_in[2];
    const float* bm1  = (const float*)d_in[3];
    const float* gm   = (const float*)d_in[4];
    const float* bmln = (const float*)d_in[5];
    const float* Wm2  = (const float*)d_in[6];
    const float* bm2  = (const float*)d_in[7];
    const float* Wu1  = (const float*)d_in[8];
    const float* bu1  = (const float*)d_in[9];
    const float* gu   = (const float*)d_in[10];
    const float* buln = (const float*)d_in[11];
    const float* Wu2  = (const float*)d_in[12];
    const float* bu2  = (const float*)d_in[13];

    const long Nn = in_sizes[0] / HID;
    const long E  = in_sizes[1] / 2;
    const long R  = E + Nn;
    const int  L  = in_sizes[2] / (2 * HID * HID);
    const long szW1 = 2 * HID * HID, szW2 = HID * HID;
    const long NF = Nn * HID;

    char* ws = (char*)d_ws;
    size_t o = 0;
    float*  aggr  = (float*)(ws + o);  o += (size_t)NF * 4;
    half_t* aggrh = (half_t*)(ws + o); o += (size_t)NF * 2;
    half_t* xh    = (half_t*)(ws + o); o += (size_t)NF * 2;
    half_t* w1m   = (half_t*)(ws + o); o += (size_t)L * szW1 * 2;
    half_t* w2m   = (half_t*)(ws + o); o += (size_t)L * szW2 * 2;
    half_t* w1u   = (half_t*)(ws + o); o += (size_t)L * szW1 * 2;
    half_t* w2u   = (half_t*)(ws + o); o += (size_t)L * szW2 * 2;

    for (int l = 0; l < L; ++l) {
        swizzle_weights<<<(int)((szW1 + 255) / 256), 256, 0, stream>>>(Wm1 + l * szW1, w1m + l * szW1, 2 * HID);
        swizzle_weights<<<(int)((szW2 + 255) / 256), 256, 0, stream>>>(Wm2 + l * szW2, w2m + l * szW2, HID);
        swizzle_weights<<<(int)((szW1 + 255) / 256), 256, 0, stream>>>(Wu1 + l * szW1, w1u + l * szW1, 2 * HID);
        swizzle_weights<<<(int)((szW2 + 255) / 256), 256, 0, stream>>>(Wu2 + l * szW2, w2u + l * szW2, HID);
    }
    // x (f32) -> xh (f16): x is only ever consumed through f16 WMMA inputs
    cvt_f32_to_f16<<<(int)((NF / 4 + 255) / 256), 256, 0, stream>>>(x_in, xh, NF);

    const long blocksE = (R + 64 * WPB / 2 - 1) / 64;   // 64 rows per block
    const long blocksN = (Nn + 63) / 64;

    for (int l = 0; l < L; ++l) {
        hipMemsetAsync(aggr, 0, (size_t)NF * 4, stream);
        msg_kernel<<<(int)((R + 63) / 64), 64, 0, stream>>>(
            xh, ei, w1m + l * szW1, w2m + l * szW2,
            bm1 + l * HID, gm + l * HID, bmln + l * HID, bm2 + l * HID,
            aggr, E, R);
        cvt_f32_to_f16<<<(int)((NF / 4 + 255) / 256), 256, 0, stream>>>(aggr, aggrh, NF);
        upd_kernel<<<(int)blocksN, 64, 0, stream>>>(
            xh, aggrh, w1u + l * szW1, w2u + l * szW2,
            bu1 + l * HID, gu + l * HID, buln + l * HID, bu2 + l * HID,
            xh, (l == L - 1) ? (float*)d_out : nullptr, Nn);
        (void)blocksE;
    }
}
